// GraphCON_GCN_conv_18107582120779
// MI455X (gfx1250) — matrix operand order repaired and checked by reading, not compiled, not run on hardware
//
#include <hip/hip_runtime.h>
#include <hip/hip_bf16.h>

typedef __attribute__((ext_vector_type(16))) _Float16 v16h;
typedef __attribute__((ext_vector_type(8)))  float    v8f;

#define NN 50000
#define EE 800000
#define NF 128
#define NH 64
#define NC 40

// ---------------------------------------------------------------------------
// degree count + rsqrt
// ---------------------------------------------------------------------------
__global__ void deg_kernel(const int* __restrict__ ei, float* __restrict__ deg, int E_) {
    int e = blockIdx.x * blockDim.x + threadIdx.x;
    if (e < E_) atomicAdd(&deg[ei[E_ + e]], 1.0f);
}

__global__ void dinv_kernel(float* __restrict__ deg, int N_) {
    int i = blockIdx.x * blockDim.x + threadIdx.x;
    if (i < N_) deg[i] = rsqrtf(deg[i] + 1.0f);
}

// ---------------------------------------------------------------------------
// Encoder: X = relu(x @ enc_W + enc_b)   [N,128] @ [128,64]
// one 16-row tile per wave, K chunks of 32, 4 N-tiles, WMMA f16->f32
// ---------------------------------------------------------------------------
__global__ __launch_bounds__(128) void enc_kernel(
    const float* __restrict__ x, const float* __restrict__ W,
    const float* __restrict__ bias, float* __restrict__ X, int N_)
{
    __shared__ _Float16 Wl[NF * NH];                       // 16 KB
    for (int i = threadIdx.x; i < NF * NH; i += blockDim.x) Wl[i] = (_Float16)W[i];
    __syncthreads();

    int wave = threadIdx.x >> 5, lane = threadIdx.x & 31;
    int tile = blockIdx.x * 4 + wave;
    int ntiles = (N_ + 15) / 16;
    if (tile >= ntiles) return;
    int row0 = tile * 16;
    int m = lane & 15, half = lane >> 4;
    int row = row0 + m; if (row >= N_) row = N_ - 1;
    const float* xr = x + (size_t)row * NF;

    v8f acc[4] = {};
#pragma unroll
    for (int kc = 0; kc < NF / 32; ++kc) {
        int b0 = kc * 32 + half * 8, b1 = kc * 32 + 16 + half * 8;
        v16h a;
#pragma unroll
        for (int i = 0; i < 8; ++i) a[i]     = (_Float16)xr[b0 + i];
#pragma unroll
        for (int i = 0; i < 8; ++i) a[i + 8] = (_Float16)xr[b1 + i];
#pragma unroll
        for (int nt = 0; nt < 4; ++nt) {
            const _Float16* p = Wl + (kc * 32 + half * 8) * NH + nt * 16 + (lane & 15);
            v16h bf;
#pragma unroll
            for (int i = 0; i < 16; ++i) bf[i] = p[(i < 8 ? i : i + 8) * NH];
            acc[nt] = __builtin_amdgcn_wmma_f32_16x16x32_f16(
                false, a, false, bf, (short)0, acc[nt], false, false);
        }
    }

    int n = lane & 15, mb = half * 8;
#pragma unroll
    for (int r = 0; r < 8; ++r) {
        int orow = row0 + mb + r;
        if (orow < N_) {
#pragma unroll
            for (int nt = 0; nt < 4; ++nt) {
                float v = acc[nt][r] + bias[nt * 16 + n];
                X[(size_t)orow * NH + nt * 16 + n] = v > 0.f ? v : 0.f;
            }
        }
    }
}

// ---------------------------------------------------------------------------
// Per-layer node kernel: XW = X @ conv_W ;
// NT = XW*self_norm + conv_b - (XW @ res_W) - res_b
// chained WMMA: xw tile restaged through LDS to build second A fragment
// ---------------------------------------------------------------------------
__global__ __launch_bounds__(128) void node_kernel(
    const float* __restrict__ X, const float* __restrict__ convW,
    const float* __restrict__ resW, const float* __restrict__ convb,
    const float* __restrict__ resb, const float* __restrict__ dinv,
    float* __restrict__ XW, float* __restrict__ NT, int N_)
{
    __shared__ _Float16 Wc[NH * NH];                       // 8 KB
    __shared__ _Float16 Wr[NH * NH];                       // 8 KB
    __shared__ float    stage[4][16 * NH];                 // 16 KB
    for (int i = threadIdx.x; i < NH * NH; i += blockDim.x) {
        Wc[i] = (_Float16)convW[i];
        Wr[i] = (_Float16)resW[i];
    }
    __syncthreads();

    int wave = threadIdx.x >> 5, lane = threadIdx.x & 31;
    int ntiles = (N_ + 15) / 16;
    int tile = blockIdx.x * 4 + wave;
    if (tile >= ntiles) tile = ntiles - 1;   // clamp: keep barrier uniform
    int row0 = tile * 16;
    int m = lane & 15, half = lane >> 4;
    int row = row0 + m; if (row >= N_) row = N_ - 1;
    const float* xr = X + (size_t)row * NH;

    // xw = X @ conv_W
    v8f acc[4] = {};
#pragma unroll
    for (int kc = 0; kc < 2; ++kc) {
        int b0 = kc * 32 + half * 8, b1 = kc * 32 + 16 + half * 8;
        v16h a;
#pragma unroll
        for (int i = 0; i < 8; ++i) a[i]     = (_Float16)xr[b0 + i];
#pragma unroll
        for (int i = 0; i < 8; ++i) a[i + 8] = (_Float16)xr[b1 + i];
#pragma unroll
        for (int nt = 0; nt < 4; ++nt) {
            const _Float16* p = Wc + (kc * 32 + half * 8) * NH + nt * 16 + (lane & 15);
            v16h bf;
#pragma unroll
            for (int i = 0; i < 16; ++i) bf[i] = p[(i < 8 ? i : i + 8) * NH];
            acc[nt] = __builtin_amdgcn_wmma_f32_16x16x32_f16(
                false, a, false, bf, (short)0, acc[nt], false, false);
        }
    }

    int n = lane & 15, mb = half * 8;
    float* sw = stage[wave];
#pragma unroll
    for (int r = 0; r < 8; ++r) {
        int orow = row0 + mb + r;
#pragma unroll
        for (int nt = 0; nt < 4; ++nt) {
            float v = acc[nt][r];
            sw[(mb + r) * NH + nt * 16 + n] = v;
            if (orow < N_) XW[(size_t)orow * NH + nt * 16 + n] = v;
        }
    }
    __syncthreads();

    // res2 = xw @ res_W  (A fragment rebuilt from LDS stage)
    v8f acc2[4] = {};
#pragma unroll
    for (int kc = 0; kc < 2; ++kc) {
        int b0 = kc * 32 + half * 8, b1 = kc * 32 + 16 + half * 8;
        v16h a;
#pragma unroll
        for (int i = 0; i < 8; ++i) a[i]     = (_Float16)sw[m * NH + b0 + i];
#pragma unroll
        for (int i = 0; i < 8; ++i) a[i + 8] = (_Float16)sw[m * NH + b1 + i];
#pragma unroll
        for (int nt = 0; nt < 4; ++nt) {
            const _Float16* p = Wr + (kc * 32 + half * 8) * NH + nt * 16 + (lane & 15);
            v16h bf;
#pragma unroll
            for (int i = 0; i < 16; ++i) bf[i] = p[(i < 8 ? i : i + 8) * NH];
            acc2[nt] = __builtin_amdgcn_wmma_f32_16x16x32_f16(
                false, a, false, bf, (short)0, acc2[nt], false, false);
        }
    }

    // NT = xw*self_norm + conv_b - res2 - res_b
#pragma unroll
    for (int r = 0; r < 8; ++r) {
        int orow = row0 + mb + r;
        int cr = orow < N_ ? orow : 0;
        float dv = dinv[cr];
        float sn = dv * dv;
        if (orow < N_) {
#pragma unroll
            for (int nt = 0; nt < 4; ++nt) {
                int col = nt * 16 + n;
                NT[(size_t)orow * NH + col] =
                    acc[nt][r] * sn + convb[col] - acc2[nt][r] - resb[col];
            }
        }
    }
}

// ---------------------------------------------------------------------------
// Fused edge kernel: 16 edges per wave.
// h = relu((X[src]-X[dst]) @ W_mlp) * X[dst]   -> atomicAdd AX3[dst]
// XW[src]*dinv[src]*dinv[dst]                  -> atomicAdd AGG[dst]
// ---------------------------------------------------------------------------
__global__ __launch_bounds__(128) void edge_kernel(
    const int* __restrict__ ei, const float* __restrict__ X,
    const float* __restrict__ XW, const float* __restrict__ dinv,
    const float* __restrict__ Wmlp, float* __restrict__ AX3,
    float* __restrict__ AGG, int E_)
{
    __shared__ _Float16 Wl[NH * NH];                       // 8 KB
    for (int i = threadIdx.x; i < NH * NH; i += blockDim.x) Wl[i] = (_Float16)Wmlp[i];
    __syncthreads();

    int wave = threadIdx.x >> 5, lane = threadIdx.x & 31;
    int tile = blockIdx.x * 4 + wave;
    int e0 = tile * 16;
    if (e0 >= E_) return;
    int m = lane & 15, half = lane >> 4;
    int e = e0 + m; if (e >= E_) e = E_ - 1;

    __builtin_prefetch(ei + e0 + 64, 0, 0);  // next tile's indices -> global_prefetch

    int src = ei[e], dst = ei[E_ + e];
    const float* Xs = X + (size_t)src * NH;
    const float* Xd = X + (size_t)dst * NH;

    v8f acc[4] = {};
#pragma unroll
    for (int kc = 0; kc < 2; ++kc) {
        int b0 = kc * 32 + half * 8, b1 = kc * 32 + 16 + half * 8;
        v16h a;
#pragma unroll
        for (int i = 0; i < 8; ++i) a[i]     = (_Float16)(Xs[b0 + i] - Xd[b0 + i]);
#pragma unroll
        for (int i = 0; i < 8; ++i) a[i + 8] = (_Float16)(Xs[b1 + i] - Xd[b1 + i]);
#pragma unroll
        for (int nt = 0; nt < 4; ++nt) {
            const _Float16* p = Wl + (kc * 32 + half * 8) * NH + nt * 16 + (lane & 15);
            v16h bf;
#pragma unroll
            for (int i = 0; i < 16; ++i) bf[i] = p[(i < 8 ? i : i + 8) * NH];
            acc[nt] = __builtin_amdgcn_wmma_f32_16x16x32_f16(
                false, a, false, bf, (short)0, acc[nt], false, false);
        }
    }

    int n = lane & 15, mb = half * 8;
#pragma unroll
    for (int r = 0; r < 8; ++r) {
        int em = e0 + mb + r;
        if (em < E_) {
            int s2 = ei[em], d2 = ei[E_ + em];
            float en = dinv[s2] * dinv[d2];
            const float* xd2 = X  + (size_t)d2 * NH;
            const float* xw2 = XW + (size_t)s2 * NH;
            float* ax = AX3 + (size_t)d2 * NH;
            float* ag = AGG + (size_t)d2 * NH;
#pragma unroll
            for (int nt = 0; nt < 4; ++nt) {
                int col = nt * 16 + n;
                float h = acc[nt][r];
                h = h > 0.f ? h : 0.f;
                atomicAdd(ax + col, h * xd2[col]);
                atomicAdd(ag + col, xw2[col] * en);
            }
        }
    }
}

// ---------------------------------------------------------------------------
// Update: X = relu(AGG + NT) + lamda1*AX3 ; re-zero AX3/AGG for next layer
// (Y state is algebraically eliminated: Y_new = relu(.) - X never feeds back)
// ---------------------------------------------------------------------------
__global__ void update_kernel(float* __restrict__ X, float* __restrict__ AX3,
                              float* __restrict__ AGG, const float* __restrict__ NT,
                              const float* __restrict__ lamda1, int total)
{
    int i = blockIdx.x * blockDim.x + threadIdx.x;
    if (i >= total) return;
    float conv = AGG[i] + NT[i];
    float r = conv > 0.f ? conv : 0.f;
    X[i] = r + lamda1[0] * AX3[i];
    AX3[i] = 0.f;
    AGG[i] = 0.f;
}

// ---------------------------------------------------------------------------
// Decoder: out = X @ dec_W + dec_b   [N,64] @ [64,40] (padded to 48 cols)
// ---------------------------------------------------------------------------
__global__ __launch_bounds__(128) void dec_kernel(
    const float* __restrict__ X, const float* __restrict__ W,
    const float* __restrict__ bias, float* __restrict__ out, int N_)
{
    __shared__ _Float16 Wl[NH * 48];                       // 6 KB, zero-padded
    for (int i = threadIdx.x; i < NH * 48; i += blockDim.x) {
        int k = i / 48, nn = i % 48;
        Wl[i] = (nn < NC) ? (_Float16)W[k * NC + nn] : (_Float16)0.0f;
    }
    __syncthreads();

    int wave = threadIdx.x >> 5, lane = threadIdx.x & 31;
    int tile = blockIdx.x * 4 + wave;
    int ntiles = (N_ + 15) / 16;
    if (tile >= ntiles) return;
    int row0 = tile * 16;
    int m = lane & 15, half = lane >> 4;
    int row = row0 + m; if (row >= N_) row = N_ - 1;
    const float* xr = X + (size_t)row * NH;

    v8f acc[3] = {};
#pragma unroll
    for (int kc = 0; kc < 2; ++kc) {
        int b0 = kc * 32 + half * 8, b1 = kc * 32 + 16 + half * 8;
        v16h a;
#pragma unroll
        for (int i = 0; i < 8; ++i) a[i]     = (_Float16)xr[b0 + i];
#pragma unroll
        for (int i = 0; i < 8; ++i) a[i + 8] = (_Float16)xr[b1 + i];
#pragma unroll
        for (int nt = 0; nt < 3; ++nt) {
            const _Float16* p = Wl + (kc * 32 + half * 8) * 48 + nt * 16 + (lane & 15);
            v16h bf;
#pragma unroll
            for (int i = 0; i < 16; ++i) bf[i] = p[(i < 8 ? i : i + 8) * 48];
            acc[nt] = __builtin_amdgcn_wmma_f32_16x16x32_f16(
                false, a, false, bf, (short)0, acc[nt], false, false);
        }
    }

    int n = lane & 15, mb = half * 8;
#pragma unroll
    for (int r = 0; r < 8; ++r) {
        int orow = row0 + mb + r;
        if (orow < N_) {
#pragma unroll
            for (int nt = 0; nt < 3; ++nt) {
                int col = nt * 16 + n;
                if (col < NC)
                    out[(size_t)orow * NC + col] = acc[nt][r] + bias[col];
            }
        }
    }
}

// ---------------------------------------------------------------------------
extern "C" void kernel_launch(void* const* d_in, const int* in_sizes, int n_in,
                              void* d_out, int out_size, void* d_ws, size_t ws_size,
                              hipStream_t stream) {
    const float* x      = (const float*)d_in[0];
    const int*   ei     = (const int*)d_in[1];     // [2,E]
    const float* encW   = (const float*)d_in[2];
    const float* encb   = (const float*)d_in[3];
    const float* convW  = (const float*)d_in[4];
    const float* convb  = (const float*)d_in[5];
    const float* resW   = (const float*)d_in[6];
    const float* resb   = (const float*)d_in[7];
    const float* wmlp   = (const float*)d_in[8];
    const float* lamda1 = (const float*)d_in[9];
    const float* decW   = (const float*)d_in[10];
    const float* decb   = (const float*)d_in[11];
    float* out = (float*)d_out;

    const int N_ = NN, E_ = EE;
    const size_t nodef = (size_t)N_ * NH;

    float* dinv = (float*)d_ws;        // N
    float* X    = dinv + N_;           // N*64
    float* XW   = X    + nodef;        // N*64
    float* NT   = XW   + nodef;        // N*64
    float* AX3  = NT   + nodef;        // N*64
    float* AGG  = AX3  + nodef;        // N*64

    const int ntiles = (N_ + 15) / 16;
    const int nblocks = (ntiles + 3) / 4;
    const int etiles = (E_ + 15) / 16;
    const int eblocks = (etiles + 3) / 4;

    // degree -> dinv
    hipMemsetAsync(dinv, 0, (size_t)N_ * sizeof(float), stream);
    deg_kernel<<<(E_ + 255) / 256, 256, 0, stream>>>(ei, dinv, E_);
    dinv_kernel<<<(N_ + 255) / 256, 256, 0, stream>>>(dinv, N_);

    // encoder
    enc_kernel<<<nblocks, 128, 0, stream>>>(x, encW, encb, X, N_);

    // atomic sinks zeroed once; update_kernel re-zeroes for subsequent layers
    hipMemsetAsync(AX3, 0, nodef * sizeof(float), stream);
    hipMemsetAsync(AGG, 0, nodef * sizeof(float), stream);

    for (int l = 0; l < 4; ++l) {
        node_kernel<<<nblocks, 128, 0, stream>>>(X, convW, resW, convb, resb,
                                                 dinv, XW, NT, N_);
        edge_kernel<<<eblocks, 128, 0, stream>>>(ei, X, XW, dinv, wmlp, AX3, AGG, E_);
        update_kernel<<<((int)nodef + 255) / 256, 256, 0, stream>>>(
            X, AX3, AGG, NT, lamda1, (int)nodef);
    }

    dec_kernel<<<nblocks, 128, 0, stream>>>(X, decW, decb, out, N_);
}
